// T5Attention_81518479278259
// MI455X (gfx1250) — matrix operand order repaired and checked by reading, not compiled, not run on hardware
//
#include <hip/hip_runtime.h>
#include <hip/hip_bf16.h>
#include <cstdint>

// ---------------------------------------------------------------------------
// T5 attention, CDNA5 (gfx1250, wave32).
//  - Projections: bf16x3 split-precision GEMM on v_wmma_f32_16x16x32_bf16
//    (fp32-grade accuracy, 2.7x fewer matrix instructions than f32 WMMA,
//     half the staging bytes).
//  - Attention core: flash-style, fp32 v_wmma_f32_16x16x4_f32 (small-K tiles,
//    softmax-precision-sensitive), K/V tiles double-buffered in LDS via
//    GLOBAL_LOAD_ASYNC_TO_LDS_B128 (ASYNCcnt) for direct memory->LDS DMA.
// B=2, S=2048, D=1024, H=16, hd=64.
// ---------------------------------------------------------------------------

typedef float  v2f   __attribute__((ext_vector_type(2)));
typedef float  v8f   __attribute__((ext_vector_type(8)));
typedef __bf16 v16bf __attribute__((ext_vector_type(16)));

#define B_  2
#define S_  2048
#define D_  1024
#define H_  16
#define HD_ 64

__device__ __forceinline__ v8f wmma_f32(v2f a, v2f b, v8f c) {
    return __builtin_amdgcn_wmma_f32_16x16x4_f32(
        false, a, false, b, (short)0, c, false, false);
}
__device__ __forceinline__ v8f wmma_bf16(v16bf a, v16bf b, v8f c) {
    return __builtin_amdgcn_wmma_f32_16x16x32_bf16(
        false, a, false, b, (short)0, c, false, false);
}

// Carrier for a 16x bf16 WMMA operand assembled from two 16B LDS loads.
union Frag16 {
    v16bf  v;
    float4 q[2];
};

// Direct global->LDS async copy (ASYNCcnt), 16B per lane per issue.
__device__ __forceinline__ void async_b128_to_lds(const void* gptr, void* lptr) {
    uint32_t lds  = (uint32_t)(uintptr_t)lptr;       // LDS aperture: low 32b
    uint64_t addr = (uint64_t)(uintptr_t)gptr;
    asm volatile("global_load_async_to_lds_b128 %0, %1, off"
                 :: "v"(lds), "v"(addr) : "memory");
}

// T5 relative-position bucket -> bias value (branchless; matches reference).
__device__ __forceinline__ float rel_bias_val(const float* __restrict__ rb,
                                              int qidx, int kidx, int h) {
    int rp  = kidx - qidx;
    int ret = (rp >= 0) ? 16 : 0;
    int n   = (rp >= 0) ? rp : -rp;
    float nf = (float)(n > 0 ? n : 1);
    int vl = 8 + (int)(__logf(nf * 0.125f) * 2.8853900817779268f); // 8/ln(16)
    vl = (vl < 15) ? vl : 15;
    int bucket = ret + ((n < 8) ? n : vl);
    return rb[bucket * H_ + h];
}

// ---------------------------------------------------------------------------
// Elementwise fp32 -> (hi, lo) bf16 split:  x ~= hi + lo, lo = bf16(x - hi).
// ---------------------------------------------------------------------------
__global__ void split_bf16_kernel(const float* __restrict__ src,
                                  __bf16* __restrict__ hi,
                                  __bf16* __restrict__ lo, int n) {
    int idx = blockIdx.x * blockDim.x + threadIdx.x;
    int stride = gridDim.x * blockDim.x;
    for (; idx < n; idx += stride) {
        float x  = src[idx];
        __bf16 h = (__bf16)x;
        hi[idx]  = h;
        lo[idx]  = (__bf16)(x - (float)h);
    }
}

// ---------------------------------------------------------------------------
// bf16x3 GEMM: C[M,N] = A[M,K] @ W[N,K]^T with A,W pre-split into hi/lo bf16.
// acc += Ah*Wh + Ah*Wl + Al*Wh  (fp32 accumulate; lo*lo term ~2^-32, dropped)
// Block: 128 thr = 4 waves, 32x32 block tile, one 16x16 per wave.
// LDS slabs 32 rows x 64 K of bf16, row stride 72 (->16B-aligned b128,
// conflict-free fragment reads).
// QKV==1: grid.y picks wq/wk/wv and scatters into [B,H,S,hd] fp32.
// ---------------------------------------------------------------------------
template <int QKV>
__global__ __launch_bounds__(128) void gemm_bf16x3(
    const __bf16* __restrict__ Ah, const __bf16* __restrict__ Al,
    const __bf16* __restrict__ Wh0, const __bf16* __restrict__ Wl0,
    const __bf16* __restrict__ Wh1, const __bf16* __restrict__ Wl1,
    const __bf16* __restrict__ Wh2, const __bf16* __restrict__ Wl2,
    float* __restrict__ D0, float* __restrict__ D1, float* __restrict__ D2,
    int M, int N, int K) {
    __shared__ __bf16 sAh[32 * 72], sAl[32 * 72];
    __shared__ __bf16 sBh[32 * 72], sBl[32 * 72];

    const int tid   = threadIdx.x;
    const int lane  = tid & 31;
    const int wave  = tid >> 5;
    const int waveM = wave >> 1;
    const int waveN = wave & 1;

    const int tilesN = N >> 5;
    const int m0 = (blockIdx.x / tilesN) * 32;
    const int n0 = (blockIdx.x % tilesN) * 32;

    const __bf16* Wh = Wh0;
    const __bf16* Wl = Wl0;
    float*        D  = D0;
    if (QKV) {
        if (blockIdx.y == 1)      { Wh = Wh1; Wl = Wl1; D = D1; }
        else if (blockIdx.y == 2) { Wh = Wh2; Wl = Wl2; D = D2; }
    }

    const int row  = lane & 15;
    const int hi   = lane >> 4;
    const int hi8  = hi * 8;    // A-frag K sub-offset
    const int hi16 = hi * 16;   // unused; B frags also use hi8

    v8f acc = {};

    for (int k0 = 0; k0 < K; k0 += 64) {
        // Stage 4 slabs (A hi/lo, B hi/lo), 8 bf16 (16B) per chunk.
        #pragma unroll
        for (int i = 0; i < 2; ++i) {
            int c  = tid + i * 128;   // 0..255
            int r  = c >> 3;          // row 0..31
            int c8 = c & 7;           // 8-elem column group
            *(float4*)&sAh[r * 72 + c8 * 8] =
                *(const float4*)&Ah[(size_t)(m0 + r) * K + k0 + c8 * 8];
            *(float4*)&sAl[r * 72 + c8 * 8] =
                *(const float4*)&Al[(size_t)(m0 + r) * K + k0 + c8 * 8];
            *(float4*)&sBh[r * 72 + c8 * 8] =
                *(const float4*)&Wh[(size_t)(n0 + r) * K + k0 + c8 * 8];
            *(float4*)&sBl[r * 72 + c8 * 8] =
                *(const float4*)&Wl[(size_t)(n0 + r) * K + k0 + c8 * 8];
        }
        __syncthreads();

        const __bf16* pah = &sAh[(waveM * 16 + row) * 72];
        const __bf16* pal = &sAl[(waveM * 16 + row) * 72];
        const __bf16* pbh = &sBh[(waveN * 16 + row) * 72];
        const __bf16* pbl = &sBl[(waveN * 16 + row) * 72];

        #pragma unroll
        for (int s = 0; s < 2; ++s) {
            const int ks = s * 32;
            Frag16 fah, fal, fbh, fbl;
            // A 16-bit layout: lanes 0-15 K={0..7,16..23}, lanes 16-31 +8
            fah.q[0] = *(const float4*)&pah[ks + hi8];
            fah.q[1] = *(const float4*)&pah[ks + hi8 + 16];
            fal.q[0] = *(const float4*)&pal[ks + hi8];
            fal.q[1] = *(const float4*)&pal[ks + hi8 + 16];
            // B 16-bit layout: lanes 0-15 K={0..7,16..23}, lanes 16-31 +8
            fbh.q[0] = *(const float4*)&pbh[ks + hi8];
            fbh.q[1] = *(const float4*)&pbh[ks + hi8 + 16];
            fbl.q[0] = *(const float4*)&pbl[ks + hi8];
            fbl.q[1] = *(const float4*)&pbl[ks + hi8 + 16];

            acc = wmma_bf16(fah.v, fbh.v, acc);
            acc = wmma_bf16(fah.v, fbl.v, acc);
            acc = wmma_bf16(fal.v, fbh.v, acc);
        }
        __syncthreads();
    }

    const int colN = n0 + waveN * 16 + (lane & 15);
    #pragma unroll
    for (int r = 0; r < 8; ++r) {
        int mm = m0 + waveM * 16 + r + 8 * hi;
        if (QKV) {
            int b_ = mm >> 11, s_ = mm & (S_ - 1);
            int h_ = colN >> 6, d_ = colN & (HD_ - 1);
            D[(((size_t)(b_ * H_ + h_) * S_ + s_) << 6) + d_] = acc[r];
        } else {
            D[(size_t)mm * N + colN] = acc[r];
        }
    }
}

// ---------------------------------------------------------------------------
// Fused flash attention, fp32 WMMA. One wave = one 16-row Q tile of one (b,h).
// Block = 64 threads = 2 waves; per-wave LDS slice with double-buffered K/V
// tiles filled by async global->LDS DMA.
// Epilogue writes the attention output pre-split (hi/lo bf16) for the
// bf16x3 output projection.
// ---------------------------------------------------------------------------
#define KT_STRIDE 68
#define PT_STRIDE 18
#define BUF_F     (2 * 16 * KT_STRIDE)                  // K+V, one buffer
#define SLICE_F   (2 * BUF_F + 16 * PT_STRIDE)          // 4640 floats

__global__ __launch_bounds__(64) void attn_flash_wmma(
    const float* __restrict__ Q, const float* __restrict__ K,
    const float* __restrict__ V, const float* __restrict__ rb,
    __bf16* __restrict__ AOh, __bf16* __restrict__ AOl) {
    __shared__ float smem[2 * SLICE_F];

    const int lane  = threadIdx.x & 31;
    const int wslot = threadIdx.x >> 5;
    float* slice = &smem[wslot * SLICE_F];
    float* Pt    = slice + 2 * BUF_F;

    const int w  = blockIdx.x * 2 + wslot;  // 0..4095
    const int qt = w & 127;
    const int bh = w >> 7;
    const int h  = bh & (H_ - 1);
    const size_t base = (size_t)bh * S_ * HD_;

    const int row = lane & 15;
    const int hi  = lane >> 4;
    const int kk  = hi << 1;

    // Q fragments (A-layout), 16 K-steps over hd=64.
    v2f qa[16];
    {
        const float* qrow = Q + base + (size_t)(qt * 16 + row) * HD_;
        #pragma unroll
        for (int t = 0; t < 16; ++t)
            qa[t] = *(const v2f*)&qrow[t * 4 + kk];
    }

    v8f o0 = {}, o1 = {}, o2 = {}, o3 = {};
    float mrun[8], lrun[8];
    #pragma unroll
    for (int r = 0; r < 8; ++r) { mrun[r] = -3.0e38f; lrun[r] = 0.0f; }

    // Async-stage one 16x64 K tile + V tile into buffer p (16 b128 issues).
    auto stage = [&](int j, int p) {
        const float* kg = K + base + (size_t)j * 16 * HD_;
        const float* vg = V + base + (size_t)j * 16 * HD_;
        float* Kt = slice + p * BUF_F;
        float* Vt = Kt + 16 * KT_STRIDE;
        #pragma unroll
        for (int i = 0; i < 8; ++i) {
            int c  = lane + i * 32;        // 0..255 float4 chunks
            int r4 = c >> 4, c4 = c & 15;
            async_b128_to_lds(kg + c * 4, &Kt[r4 * KT_STRIDE + c4 * 4]);
            async_b128_to_lds(vg + c * 4, &Vt[r4 * KT_STRIDE + c4 * 4]);
        }
    };

    stage(0, 0);

    const int NT = S_ / 16;  // 128 key tiles
    for (int j = 0; j < NT; ++j) {
        const int cur = j & 1;
        float* Kt = slice + cur * BUF_F;
        float* Vt = Kt + 16 * KT_STRIDE;

        if (j + 1 < NT) {
            // WAR guard: all LDS reads of the buffer being refilled are done.
            asm volatile("s_wait_dscnt 0" ::: "memory");
            stage(j + 1, cur ^ 1);
            // Tile j ready once only tile j+1's 16 issues remain in flight.
            asm volatile("s_wait_asynccnt 0x10" ::: "memory");
        } else {
            asm volatile("s_wait_asynccnt 0x0" ::: "memory");
        }

        // S = Q @ K^T  (B[k][n] = K[j*16+n][k])
        v8f sc = {};
        #pragma unroll
        for (int t = 0; t < 16; ++t) {
            v2f b = *(const v2f*)&Kt[(lane & 15) * KT_STRIDE + t * 4 + kk];
            sc = wmma_f32(qa[t], b, sc);
        }

        // scale + T5 bias; online softmax (row stats via 16-lane butterflies)
        const int kcol = j * 16 + (lane & 15);
        float p[8];
        #pragma unroll
        for (int r = 0; r < 8; ++r) {
            int qidx = qt * 16 + r + 8 * hi;
            float sv = sc[r] * 0.125f + rel_bias_val(rb, qidx, kcol, h);

            float tm = sv;
            tm = fmaxf(tm, __shfl_xor(tm, 1));
            tm = fmaxf(tm, __shfl_xor(tm, 2));
            tm = fmaxf(tm, __shfl_xor(tm, 4));
            tm = fmaxf(tm, __shfl_xor(tm, 8));
            float mn = fmaxf(mrun[r], tm);

            float pe = __expf(sv - mn);
            float rs = pe;
            rs += __shfl_xor(rs, 1);
            rs += __shfl_xor(rs, 2);
            rs += __shfl_xor(rs, 4);
            rs += __shfl_xor(rs, 8);

            float alpha = __expf(mrun[r] - mn);
            lrun[r] = lrun[r] * alpha + rs;
            mrun[r] = mn;
            p[r] = pe;
            o0[r] *= alpha; o1[r] *= alpha; o2[r] *= alpha; o3[r] *= alpha;
        }

        // P: C-layout -> LDS -> A-layout
        #pragma unroll
        for (int r = 0; r < 8; ++r)
            Pt[(r + 8 * hi) * PT_STRIDE + (lane & 15)] = p[r];
        asm volatile("s_wait_dscnt 0" ::: "memory");

        // O += P @ V
        #pragma unroll
        for (int t = 0; t < 4; ++t) {
            v2f a = *(const v2f*)&Pt[row * PT_STRIDE + t * 4 + kk];
            const int kr = t * 4 + kk;
            const int cl = lane & 15;
            v2f b0, b1, b2, b3;
            b0.x = Vt[kr * KT_STRIDE + cl];
            b0.y = Vt[(kr + 1) * KT_STRIDE + cl];
            b1.x = Vt[kr * KT_STRIDE + 16 + cl];
            b1.y = Vt[(kr + 1) * KT_STRIDE + 16 + cl];
            b2.x = Vt[kr * KT_STRIDE + 32 + cl];
            b2.y = Vt[(kr + 1) * KT_STRIDE + 32 + cl];
            b3.x = Vt[kr * KT_STRIDE + 48 + cl];
            b3.y = Vt[(kr + 1) * KT_STRIDE + 48 + cl];
            o0 = wmma_f32(a, b0, o0);
            o1 = wmma_f32(a, b1, o1);
            o2 = wmma_f32(a, b2, o2);
            o3 = wmma_f32(a, b3, o3);
        }
    }

    // Epilogue: normalize and store pre-split hi/lo bf16 into [B,S,D].
    const int b_ = bh >> 4;
    #pragma unroll
    for (int r = 0; r < 8; ++r) {
        float inv = 1.0f / lrun[r];
        int s_ = qt * 16 + r + 8 * hi;
        size_t idx =
            ((size_t)(b_ * S_) + s_) * D_ + h * HD_ + (lane & 15);
        float vals[4] = {o0[r] * inv, o1[r] * inv, o2[r] * inv, o3[r] * inv};
        #pragma unroll
        for (int d = 0; d < 4; ++d) {
            float  x = vals[d];
            __bf16 hbits = (__bf16)x;
            AOh[idx + d * 16] = hbits;
            AOl[idx + d * 16] = (__bf16)(x - (float)hbits);
        }
    }
}

// ---------------------------------------------------------------------------
// past_bias output: [1,H,S,S].
// ---------------------------------------------------------------------------
__global__ void bias_fill(const float* __restrict__ rb, float* __restrict__ out) {
    const long long total = (long long)H_ * S_ * S_;
    long long idx = (long long)blockIdx.x * blockDim.x + threadIdx.x;
    const long long stride = (long long)gridDim.x * blockDim.x;
    for (; idx < total; idx += stride) {
        int h   = (int)(idx >> 22);
        int rem = (int)(idx & ((1 << 22) - 1));
        int q = rem >> 11;
        int k = rem & (S_ - 1);
        out[idx] = rel_bias_val(rb, q, k, h);
    }
}

// ---------------------------------------------------------------------------
extern "C" void kernel_launch(void* const* d_in, const int* in_sizes, int n_in,
                              void* d_out, int out_size, void* d_ws,
                              size_t ws_size, hipStream_t stream) {
    const float* x  = (const float*)d_in[0];
    const float* wq = (const float*)d_in[1];
    const float* wk = (const float*)d_in[2];
    const float* wv = (const float*)d_in[3];
    const float* wo = (const float*)d_in[4];
    const float* rb = (const float*)d_in[5];

    float* out = (float*)d_out;

    // Workspace carve-out (bytes).
    const size_t NX = (size_t)B_ * S_ * D_;   // 4,194,304 (x / AO / each QKV)
    const size_t NW = (size_t)D_ * D_;        // 1,048,576 (each weight)
    char* p = (char*)d_ws;
    float*  Qb  = (float*)p;  p += NX * sizeof(float);
    float*  Kb  = (float*)p;  p += NX * sizeof(float);
    float*  Vb  = (float*)p;  p += NX * sizeof(float);
    __bf16* xh  = (__bf16*)p; p += NX * sizeof(__bf16);
    __bf16* xl  = (__bf16*)p; p += NX * sizeof(__bf16);
    __bf16* aoh = (__bf16*)p; p += NX * sizeof(__bf16);
    __bf16* aol = (__bf16*)p; p += NX * sizeof(__bf16);
    __bf16* wqh = (__bf16*)p; p += NW * sizeof(__bf16);
    __bf16* wql = (__bf16*)p; p += NW * sizeof(__bf16);
    __bf16* wkh = (__bf16*)p; p += NW * sizeof(__bf16);
    __bf16* wkl = (__bf16*)p; p += NW * sizeof(__bf16);
    __bf16* wvh = (__bf16*)p; p += NW * sizeof(__bf16);
    __bf16* wvl = (__bf16*)p; p += NW * sizeof(__bf16);
    __bf16* woh = (__bf16*)p; p += NW * sizeof(__bf16);
    __bf16* wol = (__bf16*)p; p += NW * sizeof(__bf16);

    // 0) fp32 -> (hi, lo) bf16 splits (one-time, bandwidth-trivial)
    split_bf16_kernel<<<4096, 256, 0, stream>>>(x,  xh,  xl,  (int)NX);
    split_bf16_kernel<<<1024, 256, 0, stream>>>(wq, wqh, wql, (int)NW);
    split_bf16_kernel<<<1024, 256, 0, stream>>>(wk, wkh, wkl, (int)NW);
    split_bf16_kernel<<<1024, 256, 0, stream>>>(wv, wvh, wvl, (int)NW);
    split_bf16_kernel<<<1024, 256, 0, stream>>>(wo, woh, wol, (int)NW);

    const int M = B_ * S_, N = D_, Kd = D_;
    const int tiles = (M / 32) * (N / 32);  // 4096

    // 1) QKV projections (bf16x3), scattered into [B,H,S,hd] fp32
    gemm_bf16x3<1><<<dim3(tiles, 3), 128, 0, stream>>>(
        xh, xl, wqh, wql, wkh, wkl, wvh, wvl, Qb, Kb, Vb, M, N, Kd);

    // 2) Flash attention (fp32 WMMA core, async K/V staging)
    attn_flash_wmma<<<(B_ * H_ * (S_ / 16)) / 2, 64, 0, stream>>>(
        Qb, Kb, Vb, rb, aoh, aol);

    // 3) Output projection (bf16x3) into d_out[0 .. B*S*D)
    gemm_bf16x3<0><<<dim3(tiles, 1), 128, 0, stream>>>(
        aoh, aol, woh, wol, woh, wol, woh, wol, out, out, out, M, N, Kd);

    // 4) past_bias into d_out[B*S*D ..)
    bias_fill<<<4096, 256, 0, stream>>>(rb, out + (size_t)B_ * S_ * D_);
}
